// Rating_att_Pre_71004399337500
// MI455X (gfx1250) — matrix operand order repaired and checked by reading, not compiled, not run on hardware
//
#include <hip/hip_runtime.h>
#include <stdint.h>
#include <math.h>

typedef __attribute__((ext_vector_type(16))) _Float16 v16h;
typedef __attribute__((ext_vector_type(8)))  float    v8f;

#define ER     500000
#define ET     250000
#define NTR    (2*ET)
#define NPADK  (1<<19)      // 524288 >= NTR
#define CHUNK  2048         // elements per LDS bitonic chunk
#define NCHUNK (NPADK / CHUNK)
#define EPSV   1e-10f

__device__ __forceinline__ float dot4f(float4 a, float4 b) {
  return a.x*b.x + a.y*b.y + a.z*b.z + a.w*b.w;
}
__device__ __forceinline__ float sigmoidf(float x) {
  return 1.0f / (1.0f + __expf(-x));
}

template <typename T>
__device__ __forceinline__ void block_reduce_atomic(T val, T* target) {
  __shared__ T sh[256];
  int tid = threadIdx.x;
  sh[tid] = val;
  __syncthreads();
  for (int s = 128; s > 0; s >>= 1) {
    if (tid < s) sh[tid] += sh[tid + s];
    __syncthreads();
  }
  if (tid == 0) atomicAdd(target, sh[0]);
}

// ---------------------------------------------------------------------------
// Init: zero accumulators, pad sort keys with max sentinels
// ---------------------------------------------------------------------------
__global__ void k_init(uint64_t* __restrict__ ka, uint64_t* __restrict__ kd,
                       float* a_latt, float* a_trust,
                       unsigned long long* a_rank, double* a_ap) {
  int i = blockIdx.x * blockDim.x + threadIdx.x;
  if (i == 0) { *a_latt = 0.f; *a_trust = 0.f; *a_rank = 0ull; *a_ap = 0.0; }
  int j = NTR + i;
  if (j < NPADK) { ka[j] = ~0ull; kd[j] = ~0ull; }
}

// ---------------------------------------------------------------------------
// Weight collapse stage 1: W45 = W4[256x128] * W5[128x64] via WMMA f16
// ---------------------------------------------------------------------------
__global__ __launch_bounds__(32)
void k_collapse_w45(const float* __restrict__ W4, const float* __restrict__ W5,
                    float* __restrict__ W45) {
  const int l  = threadIdx.x;
  const int r  = l & 15;
  const int hi = l >> 4;
  const int m0 = (blockIdx.x >> 2) << 4;   // 256/16 = 16 tiles
  const int n0 = (blockIdx.x & 3) << 4;    //  64/16 =  4 tiles
  v8f acc = {0.f,0.f,0.f,0.f,0.f,0.f,0.f,0.f};
  for (int k0 = 0; k0 < 128; k0 += 32) {
    v16h a, b;
#pragma unroll
    for (int e = 0; e < 16; ++e) {
      // A (16xK f16) lane layout: K = (e&7) + (e>>3)*16 + hi*8
      int ka = k0 + (e & 7) + ((e >> 3) << 4) + (hi << 3);
      a[e] = (_Float16)W4[(m0 + r) * 128 + ka];
      // B (Kx16 f16) lane layout: K = (e&7) + (e>>3)*16 + hi*8, N = r
      int kb = k0 + (e & 7) + ((e >> 3) << 4) + (hi << 3);
      b[e] = (_Float16)W5[kb * 64 + (n0 + r)];
    }
    acc = __builtin_amdgcn_wmma_f32_16x16x32_f16(false, a, false, b,
                                                 (short)0, acc, false, false);
  }
#pragma unroll
  for (int v = 0; v < 8; ++v)  // D: M = hi*8 + v, N = r
    W45[(m0 + (hi << 3) + v) * 64 + (n0 + r)] = acc[v];
}

// ---------------------------------------------------------------------------
// Weight collapse stage 2: w_influ = W45*W6, w_trust = Wso*Wso2, bias consts
// ---------------------------------------------------------------------------
__global__ __launch_bounds__(256)
void k_collapse_final(const float* __restrict__ W45, const float* __restrict__ W5,
                      const float* __restrict__ W6,
                      const float* __restrict__ W4b, const float* __restrict__ W5b,
                      const float* __restrict__ W6b,
                      const float* __restrict__ Wso, const float* __restrict__ Wso2,
                      const float* __restrict__ Wsob, const float* __restrict__ Wso2b,
                      float* __restrict__ w_influ, float* __restrict__ w_trust,
                      float* __restrict__ consts) {
  __shared__ float v5[64];
  int tid = threadIdx.x;
  {
    float s = 0.f;
    for (int n = 0; n < 64; ++n) s += W45[tid * 64 + n] * W6[n];
    w_influ[tid] = s;
  }
  if (tid < 128) {
    float s = 0.f;
    for (int n = 0; n < 64; ++n) s += Wso[tid * 64 + n] * Wso2[n];
    w_trust[tid] = s;
  }
  if (tid < 64) {
    float s = W5b[tid];
    for (int k = 0; k < 128; ++k) s += W4b[k] * W5[k * 64 + tid];
    v5[tid] = s;
  }
  __syncthreads();
  if (tid == 0) {
    float ci = W6b[0];
    for (int n = 0; n < 64; ++n) ci += v5[n] * W6[n];
    consts[0] = ci;                                  // c_influ
    float ct = Wso2b[0];
    for (int n = 0; n < 64; ++n) ct += Wsob[n] * Wso2[n];
    consts[1] = ct;                                  // c_trust
  }
}

// ---------------------------------------------------------------------------
// Rated edges: att (sigmoid 128-dot) + influ (256-dot) ; 16 lanes per edge
// ---------------------------------------------------------------------------
__global__ __launch_bounds__(256)
void k_rated(const float4* __restrict__ x_u, const float4* __restrict__ x_i,
             const float4* __restrict__ a_u, const float4* __restrict__ a_i,
             const float4* __restrict__ ui,  const float4* __restrict__ ii,
             const float* __restrict__ boun3,
             const float4* __restrict__ Watt4, const float* __restrict__ Wattb,
             const int* __restrict__ src, const int* __restrict__ dst,
             const float4* __restrict__ wi4, const float* __restrict__ consts,
             float* __restrict__ out, float* __restrict__ a_latt) {
  const int tid = threadIdx.x;
  const int e   = blockIdx.x * 16 + (tid >> 4);
  const int r   = tid & 15;
  float partial = 0.f;
  if (e < ER) {
    const int s = src[e], d = dst[e];
    // influ feature order: item_intent[d] | x_u[s] | x_i[d] | user_intent[s]
    float inf = dot4f(ii[d * 16 + r],  wi4[r])
              + dot4f(x_u[s * 16 + r], wi4[16 + r])
              + dot4f(x_i[d * 16 + r], wi4[32 + r])
              + dot4f(ui[s * 16 + r],  wi4[48 + r]);
    float att = dot4f(a_u[s * 16 + r], Watt4[r])
              + dot4f(a_i[d * 16 + r], Watt4[16 + r]);
#pragma unroll
    for (int m = 8; m > 0; m >>= 1) {
      inf += __shfl_xor(inf, m, 16);
      att += __shfl_xor(att, m, 16);
    }
    if (r == 0) {
      float attv = sigmoidf(att + Wattb[0]);
      float influ = inf + consts[0];
      // score = a_sim*(att>=.5 ? +0 : -0); a_sim>0 so score = signed zero
      out[e] = influ + (attv >= 0.5f ? 0.0f : -0.0f);
      float dd = boun3[e] - attv + EPSV;
      partial = dd * dd;
    }
  }
  block_reduce_atomic<float>(partial, a_latt);
}

// ---------------------------------------------------------------------------
// Trust edges: score = sigmoid(128-dot + c); build stable sort keys
// ---------------------------------------------------------------------------
__global__ __launch_bounds__(256)
void k_trust(const float4* __restrict__ t,
             const int* __restrict__ tps, const int* __restrict__ tpd,
             const int* __restrict__ tns, const int* __restrict__ tnd,
             const float4* __restrict__ wt4, const float* __restrict__ consts,
             uint64_t* __restrict__ ka, uint64_t* __restrict__ kd,
             float* __restrict__ a_trust) {
  const int tid = threadIdx.x;
  const int e   = blockIdx.x * 16 + (tid >> 4);
  const int r   = tid & 15;
  float partial = 0.f;
  if (e < NTR) {
    const bool pos = (e < ET);
    const int  ei  = pos ? e : e - ET;
    const int  s   = pos ? tps[ei] : tns[ei];
    const int  d   = pos ? tpd[ei] : tnd[ei];
    float acc = dot4f(t[s * 16 + r], wt4[r])
              + dot4f(t[d * 16 + r], wt4[16 + r]);
#pragma unroll
    for (int m = 8; m > 0; m >>= 1) acc += __shfl_xor(acc, m, 16);
    if (r == 0) {
      float sc = sigmoidf(acc + consts[1]);
      uint32_t bits = __float_as_uint(sc);          // sc>0 -> bits monotone
      ka[e] = ((uint64_t)bits << 20) | (uint32_t)e;           // asc score, asc idx
      kd[e] = ((uint64_t)(~bits) << 20) | (uint32_t)e;        // desc score, asc idx
      float dd = (pos ? 1.0f : 0.0f) - sc + EPSV;
      partial = dd * dd;
    }
  }
  block_reduce_atomic<float>(partial, a_trust);
}

// ---------------------------------------------------------------------------
// Bitonic sort: LDS-fused chunks of 2048 + global passes for j >= CHUNK.
// Direction always computed from the GLOBAL element index: up = ((gi&k)==0).
// ---------------------------------------------------------------------------
__device__ __forceinline__ void bitonic_lds_pass(uint64_t* ls, unsigned base,
                                                 unsigned j, unsigned k) {
  // pairs: insert a 0 at bit log2(j): i = ((p & ~(j-1)) << 1) | (p & (j-1))
  for (unsigned p = threadIdx.x; p < CHUNK / 2; p += 256) {
    unsigned i   = ((p & ~(j - 1)) << 1) | (p & (j - 1));
    unsigned ixj = i | j;
    uint64_t a = ls[i], b = ls[ixj];
    bool up = (((base + i) & k) == 0);
    if ((a > b) == up) { ls[i] = b; ls[ixj] = a; }
  }
}

// Full bitonic prefix (k = 2 .. CHUNK) inside one LDS chunk.
__global__ __launch_bounds__(256)
void k_bitonic_local_full(uint64_t* __restrict__ ka, uint64_t* __restrict__ kd) {
  __shared__ uint64_t ls[CHUNK];
  const int  b    = blockIdx.x;
  uint64_t*  data = (b >= NCHUNK) ? kd : ka;
  const unsigned cb   = (b >= NCHUNK) ? (b - NCHUNK) : b;
  const unsigned base = cb * CHUNK;
#pragma unroll
  for (int q = 0; q < CHUNK / 256; ++q)
    ls[threadIdx.x + q * 256] = data[base + threadIdx.x + q * 256];
  __syncthreads();
  for (unsigned k = 2; k <= CHUNK; k <<= 1)
    for (unsigned j = k >> 1; j > 0; j >>= 1) {
      bitonic_lds_pass(ls, base, j, k);
      __syncthreads();
    }
#pragma unroll
  for (int q = 0; q < CHUNK / 256; ++q)
    data[base + threadIdx.x + q * 256] = ls[threadIdx.x + q * 256];
}

// Tail passes (j = CHUNK/2 .. 1) of outer stage k, inside one LDS chunk.
__global__ __launch_bounds__(256)
void k_bitonic_local_tail(uint64_t* __restrict__ ka, uint64_t* __restrict__ kd,
                          unsigned k) {
  __shared__ uint64_t ls[CHUNK];
  const int  b    = blockIdx.x;
  uint64_t*  data = (b >= NCHUNK) ? kd : ka;
  const unsigned cb   = (b >= NCHUNK) ? (b - NCHUNK) : b;
  const unsigned base = cb * CHUNK;
#pragma unroll
  for (int q = 0; q < CHUNK / 256; ++q)
    ls[threadIdx.x + q * 256] = data[base + threadIdx.x + q * 256];
  __syncthreads();
  for (unsigned j = CHUNK >> 1; j > 0; j >>= 1) {
    bitonic_lds_pass(ls, base, j, k);
    __syncthreads();
  }
#pragma unroll
  for (int q = 0; q < CHUNK / 256; ++q)
    data[base + threadIdx.x + q * 256] = ls[threadIdx.x + q * 256];
}

// Global pass for j >= CHUNK (partner crosses chunk boundary).
__global__ __launch_bounds__(256)
void k_bitonic_global(uint64_t* __restrict__ ka, uint64_t* __restrict__ kd,
                      unsigned j, unsigned k) {
  unsigned gi = blockIdx.x * blockDim.x + threadIdx.x;
  uint64_t* data = (gi >= NPADK) ? kd : ka;
  unsigned i   = gi & (NPADK - 1);
  unsigned ixj = i ^ j;
  if (ixj > i) {
    uint64_t a = data[i], b = data[ixj];
    bool up = ((i & k) == 0);
    if ((a > b) == up) { data[i] = b; data[ixj] = a; }
  }
}

// AUC: sum of 1-based ranks of positives in ascending order
__global__ __launch_bounds__(256)
void k_auc(const uint64_t* __restrict__ ka, unsigned long long* a_rank) {
  int i = blockIdx.x * 256 + threadIdx.x;
  unsigned long long v = 0ull;
  if (i < NTR && (unsigned)(ka[i] & 0xFFFFFu) < ET)
    v = (unsigned long long)(i + 1);
  block_reduce_atomic<unsigned long long>(v, a_rank);
}

// AP scan stage 1: per-block inclusive scan of positive flags (desc order)
__global__ __launch_bounds__(256)
void k_scan1(const uint64_t* __restrict__ kd, uint32_t* __restrict__ scan,
             uint32_t* __restrict__ blk) {
  __shared__ uint32_t sh[256];
  int i = blockIdx.x * 256 + threadIdx.x;
  uint32_t f = (i < NTR && (unsigned)(kd[i] & 0xFFFFFu) < ET) ? 1u : 0u;
  sh[threadIdx.x] = f;
  __syncthreads();
  for (int off = 1; off < 256; off <<= 1) {
    uint32_t v = (threadIdx.x >= off) ? sh[threadIdx.x - off] : 0u;
    __syncthreads();
    sh[threadIdx.x] += v;
    __syncthreads();
  }
  if (i < NTR) scan[i] = sh[threadIdx.x];
  if (threadIdx.x == 255) blk[blockIdx.x] = sh[255];
}

// AP scan stage 2: serial exclusive scan of block sums (tiny)
__global__ void k_scan2(uint32_t* __restrict__ blk, int nblk) {
  if (threadIdx.x == 0 && blockIdx.x == 0) {
    uint32_t run = 0;
    for (int b = 0; b < nblk; ++b) { uint32_t v = blk[b]; blk[b] = run; run += v; }
  }
}

// AP stage 3: sum_{positives} k / rank
__global__ __launch_bounds__(256)
void k_ap(const uint64_t* __restrict__ kd, const uint32_t* __restrict__ scan,
          const uint32_t* __restrict__ blk, double* a_ap) {
  int i = blockIdx.x * 256 + threadIdx.x;
  double term = 0.0;
  if (i < NTR && (unsigned)(kd[i] & 0xFFFFFu) < ET) {
    uint32_t kk = scan[i] + blk[blockIdx.x];
    term = (double)kk / (double)(i + 1);
  }
  block_reduce_atomic<double>(term, a_ap);
}

__global__ void k_final(const float* a_latt, const float* a_trust,
                        const unsigned long long* a_rank, const double* a_ap,
                        float* __restrict__ out) {
  if (threadIdx.x == 0 && blockIdx.x == 0) {
    out[ER]     = sqrtf(*a_latt / (float)ER);
    double P = (double)ET, Nn = (double)ET;
    out[ER + 1] = (float)(((double)*a_rank - P * (P + 1.0) * 0.5) / (P * Nn));
    out[ER + 2] = (float)(*a_ap / P);
    out[ER + 3] = sqrtf(*a_trust / (float)NTR);
  }
}

// ---------------------------------------------------------------------------
extern "C" void kernel_launch(void* const* d_in, const int* in_sizes, int n_in,
                              void* d_out, int out_size, void* d_ws, size_t ws_size,
                              hipStream_t stream) {
  const float* x_u  = (const float*)d_in[0];
  const float* x_i  = (const float*)d_in[1];
  const float* a_u  = (const float*)d_in[2];
  const float* a_i  = (const float*)d_in[3];
  const float* t    = (const float*)d_in[4];
  const float* boun3= (const float*)d_in[5];
  const float* ui   = (const float*)d_in[6];
  const float* ii   = (const float*)d_in[7];
  const float* Watt = (const float*)d_in[8];
  const float* Wattb= (const float*)d_in[9];
  const float* W4   = (const float*)d_in[10];
  const float* W4b  = (const float*)d_in[11];
  const float* W5   = (const float*)d_in[12];
  const float* W5b  = (const float*)d_in[13];
  const float* W6   = (const float*)d_in[14];
  const float* W6b  = (const float*)d_in[15];
  const float* Wso  = (const float*)d_in[16];
  const float* Wsob = (const float*)d_in[17];
  const float* Wso2 = (const float*)d_in[18];
  const float* Wso2b= (const float*)d_in[19];
  const int* rsrc = (const int*)d_in[20];
  const int* rdst = (const int*)d_in[21];
  const int* tps  = (const int*)d_in[22];
  const int* tpd  = (const int*)d_in[23];
  const int* tns  = (const int*)d_in[24];
  const int* tnd  = (const int*)d_in[25];
  float* out = (float*)d_out;

  char* ws = (char*)d_ws;
  float*    W45      = (float*)(ws + 0);                  // 256*64*4 = 65536
  float*    w_influ  = (float*)(ws + 65536);              // 1024
  float*    w_trust  = (float*)(ws + 66560);              // 512
  float*    consts   = (float*)(ws + 67072);              // c_influ, c_trust
  float*    a_latt   = (float*)(ws + 67136);
  float*    a_trust  = (float*)(ws + 67140);
  unsigned long long* a_rank = (unsigned long long*)(ws + 67144);
  double*   a_ap     = (double*)(ws + 67152);
  uint64_t* ka   = (uint64_t*)(ws + 131072);
  uint64_t* kd   = (uint64_t*)(ws + 131072 + (size_t)NPADK * 8);
  uint32_t* scan = (uint32_t*)(ws + 131072 + (size_t)NPADK * 16);
  uint32_t* blk  = (uint32_t*)(ws + 131072 + (size_t)NPADK * 16 + (size_t)NPADK * 4);

  const int nblk = (NTR + 255) / 256;   // 1954

  k_init<<<(NPADK - NTR + 255) / 256, 256, 0, stream>>>(ka, kd, a_latt, a_trust, a_rank, a_ap);
  k_collapse_w45<<<64, 32, 0, stream>>>(W4, W5, W45);
  k_collapse_final<<<1, 256, 0, stream>>>(W45, W5, W6, W4b, W5b, W6b,
                                          Wso, Wso2, Wsob, Wso2b,
                                          w_influ, w_trust, consts);
  k_rated<<<ER / 16, 256, 0, stream>>>((const float4*)x_u, (const float4*)x_i,
                                       (const float4*)a_u, (const float4*)a_i,
                                       (const float4*)ui,  (const float4*)ii,
                                       boun3, (const float4*)Watt, Wattb,
                                       rsrc, rdst, (const float4*)w_influ, consts,
                                       out, a_latt);
  k_trust<<<NTR / 16, 256, 0, stream>>>((const float4*)t, tps, tpd, tns, tnd,
                                        (const float4*)w_trust, consts,
                                        ka, kd, a_trust);

  // Bitonic sort over both key arrays (2*NCHUNK blocks for local kernels).
  k_bitonic_local_full<<<2 * NCHUNK, 256, 0, stream>>>(ka, kd);
  for (unsigned k = 2 * CHUNK; k <= (unsigned)NPADK; k <<= 1) {
    for (unsigned j = k >> 1; j >= CHUNK; j >>= 1)
      k_bitonic_global<<<2 * NPADK / 256, 256, 0, stream>>>(ka, kd, j, k);
    k_bitonic_local_tail<<<2 * NCHUNK, 256, 0, stream>>>(ka, kd, k);
  }

  k_auc<<<nblk, 256, 0, stream>>>(ka, a_rank);
  k_scan1<<<nblk, 256, 0, stream>>>(kd, scan, blk);
  k_scan2<<<1, 1, 0, stream>>>(blk, nblk);
  k_ap<<<nblk, 256, 0, stream>>>(kd, scan, blk, a_ap);
  k_final<<<1, 1, 0, stream>>>(a_latt, a_trust, a_rank, a_ap, out);
}
